// DeformableTransformer_15204184228645
// MI455X (gfx1250) — matrix operand order repaired and checked by reading, not compiled, not run on hardware
//
#include <hip/hip_runtime.h>

// ---------------- model constants ----------------
#define BB   2
#define Cc   256
#define HHn  8
#define DHn  32
#define LLn  4
#define PPn  4
#define NQn  300
#define NLn  6
#define FFn  1024
#define STOT 13294

typedef unsigned int   u32;
typedef unsigned short u16;

__constant__ int c_HW[4] = {100, 50, 25, 13};     // square levels, H==W
__constant__ int c_ST[4] = {0, 10000, 12500, 13125};

// ---------------- WMMA types ----------------
typedef __bf16 v16bf __attribute__((ext_vector_type(16)));
typedef float  v8f   __attribute__((ext_vector_type(8)));

union FragU { u32 u[8]; v16bf v; };

__device__ inline u16 f2bfu(float f) {
    u32 u = __builtin_bit_cast(u32, f);
    return (u16)((u + 0x7FFFu + ((u >> 16) & 1u)) >> 16);   // RNE
}

// ---- CDNA5 async global->LDS copy (guarded: falls back to reg staging) ----
#if defined(__has_builtin)
# if __has_builtin(__builtin_amdgcn_global_load_async_to_lds_b128)
#  define HAVE_ASYNC_LDS 1
# endif
#endif
#ifndef HAVE_ASYNC_LDS
# define HAVE_ASYNC_LDS 0
#endif

#if HAVE_ASYNC_LDS
typedef int v4i_gcc __attribute__((vector_size(4 * sizeof(int))));
typedef __attribute__((address_space(1))) v4i_gcc gv4;   // global int4
typedef __attribute__((address_space(3))) v4i_gcc lv4;   // LDS int4
# if defined(__has_builtin) && __has_builtin(__builtin_amdgcn_s_wait_asynccnt)
#  define WAIT_ASYNC() __builtin_amdgcn_s_wait_asynccnt(0)
# else
#  define WAIT_ASYNC() asm volatile("s_wait_asynccnt 0" ::: "memory")
# endif
#endif

// ---------------- bf16 GEMM: Y[M,N] = Xb[M,K] @ Wb[N,K]^T + bias ----------------
// block = 128 threads (4 waves). Block tile 32x128; each wave: 2 row-tiles x 2 col-tiles
// = 4 v_wmma_f32_16x16x32_bf16 per K-step. Double-buffered LDS software pipeline.
// Requires N % 128 == 0, K % 32 == 0.
__global__ __launch_bounds__(128)
void k_gemm_bf16(const u16* __restrict__ Xb, const u16* __restrict__ Wb,
                 const float* __restrict__ bias, void* __restrict__ Yp,
                 int M, int N, int K, int relu, int obf)
{
    __shared__ __align__(16) u32 As[2][32][20];   // 32 rows x 16 uints, stride 20 (80B)
    __shared__ __align__(16) u32 Bs[2][128][20];  // 128 cols x 16 uints

    const int tid  = threadIdx.x;
    const int wv   = tid >> 5;
    const int lane = tid & 31;
    const int half = lane >> 4;
    const int r    = lane & 15;
    const int row0 = blockIdx.y * 32;
    const int col0 = blockIdx.x * 128;

    // A staging: thread -> (row = tid>>2, seg = tid&3), one uint4 = 8 bf16.
    // Rows past M are clamped (their garbage only feeds output rows that the
    // epilogue discards), so no zeroing needed.
    const int arow  = tid >> 2;
    const int aseg  = tid & 3;
    const int arowc = min(row0 + arow, M - 1);
    const u16* aptr = Xb + (long)arowc * K + aseg * 8;
    // B staging: thread -> one weight row (col0 + tid), 4 x uint4
    const u16* bptr = Wb + (long)(col0 + tid) * K;

    v8f acc00 = {}, acc01 = {}, acc10 = {}, acc11 = {};

    auto compute = [&](int pb) {
        FragU fa0, fa1, fb0, fb1;
        #pragma unroll
        for (int v2 = 0; v2 < 8; ++v2) {
            const int ia = ((v2 < 4) ? v2 : v2 + 4) + 4 * half;
            fa0.u[v2] = As[pb][r][ia];
            fa1.u[v2] = As[pb][16 + r][ia];
            const int ib = ((v2 < 4) ? v2 : v2 + 4) + 4 * half;
            fb0.u[v2] = Bs[pb][wv * 16 + r][ib];
            fb1.u[v2] = Bs[pb][(wv + 4) * 16 + r][ib];
        }
        acc00 = __builtin_amdgcn_wmma_f32_16x16x32_bf16(false, fa0.v, false, fb0.v, (short)0, acc00, false, false);
        acc01 = __builtin_amdgcn_wmma_f32_16x16x32_bf16(false, fa0.v, false, fb1.v, (short)0, acc01, false, false);
        acc10 = __builtin_amdgcn_wmma_f32_16x16x32_bf16(false, fa1.v, false, fb0.v, (short)0, acc10, false, false);
        acc11 = __builtin_amdgcn_wmma_f32_16x16x32_bf16(false, fa1.v, false, fb1.v, (short)0, acc11, false, false);
    };

#if HAVE_ASYNC_LDS
    auto stage = [&](int pb, int kb) {
        __builtin_amdgcn_global_load_async_to_lds_b128(
            (gv4*)(aptr + kb), (lv4*)&As[pb][arow][aseg * 4], 0, 0);
        #pragma unroll
        for (int j = 0; j < 4; ++j)
            __builtin_amdgcn_global_load_async_to_lds_b128(
                (gv4*)(bptr + kb + j * 8), (lv4*)&Bs[pb][tid][j * 4], 0, 0);
    };
    stage(0, 0);
    int p = 0;
    for (int kb = 0; kb < K; kb += 32) {
        WAIT_ASYNC();
        __syncthreads();
        if (kb + 32 < K) stage(p ^ 1, kb + 32);
        compute(p);
        p ^= 1;
    }
#else
    uint4 ra, rb0, rb1, rb2, rb3;
    auto ldregs = [&](int kb) {
        ra  = *(const uint4*)(aptr + kb);
        rb0 = *(const uint4*)(bptr + kb);
        rb1 = *(const uint4*)(bptr + kb + 8);
        rb2 = *(const uint4*)(bptr + kb + 16);
        rb3 = *(const uint4*)(bptr + kb + 24);
    };
    auto stregs = [&](int pb) {
        *(uint4*)&As[pb][arow][aseg * 4] = ra;
        *(uint4*)&Bs[pb][tid][0]  = rb0;
        *(uint4*)&Bs[pb][tid][4]  = rb1;
        *(uint4*)&Bs[pb][tid][8]  = rb2;
        *(uint4*)&Bs[pb][tid][12] = rb3;
    };
    ldregs(0);
    int p = 0;
    for (int kb = 0; kb < K; kb += 32) {
        stregs(p);
        __syncthreads();
        if (kb + 32 < K) {
            ldregs(kb + 32);
            __builtin_prefetch(bptr + kb + 64, 0, 0);
        }
        compute(p);
        p ^= 1;
    }
#endif

    const int c0 = col0 + wv * 16 + r;
    const int c1 = col0 + (wv + 4) * 16 + r;
    const float bv0 = bias ? bias[c0] : 0.0f;
    const float bv1 = bias ? bias[c1] : 0.0f;
    float* Yf = (float*)Yp;
    u16*   Yh = (u16*)Yp;

    #pragma unroll
    for (int i2 = 0; i2 < 8; ++i2) {
        int rA = row0 + i2 + 8 * half;
        int rB = rA + 16;
        float y00 = acc00[i2] + bv0, y01 = acc01[i2] + bv1;
        float y10 = acc10[i2] + bv0, y11 = acc11[i2] + bv1;
        if (relu) {
            y00 = fmaxf(y00, 0.0f); y01 = fmaxf(y01, 0.0f);
            y10 = fmaxf(y10, 0.0f); y11 = fmaxf(y11, 0.0f);
        }
        if (rA < M) {
            if (obf) { Yh[(long)rA * N + c0] = f2bfu(y00); Yh[(long)rA * N + c1] = f2bfu(y01); }
            else     { Yf[(long)rA * N + c0] = y00;        Yf[(long)rA * N + c1] = y01; }
        }
        if (rB < M) {
            if (obf) { Yh[(long)rB * N + c0] = f2bfu(y10); Yh[(long)rB * N + c1] = f2bfu(y11); }
            else     { Yf[(long)rB * N + c0] = y10;        Yf[(long)rB * N + c1] = y11; }
        }
    }
}

// ---------------- fp32 -> bf16 bulk convert ----------------
__global__ void k_cvt(const float* __restrict__ x, u16* __restrict__ y, long n)
{
    long i = (blockIdx.x * (long)blockDim.x + threadIdx.x) * 4;
    if (i + 3 < n) {
        float4 v = *(const float4*)(x + i);
        y[i]     = f2bfu(v.x);
        y[i + 1] = f2bfu(v.y);
        y[i + 2] = f2bfu(v.z);
        y[i + 3] = f2bfu(v.w);
    } else {
        for (long j = i; j < n; ++j) y[j] = f2bfu(x[j]);
    }
}

// ---------------- fused adds producing bf16 GEMM inputs ----------------
__global__ void k_add_bf16(const float* __restrict__ a, const float* __restrict__ b,
                           u16* __restrict__ o, long n)
{
    long i = blockIdx.x * (long)blockDim.x + threadIdx.x;
    if (i < n) o[i] = f2bfu(a[i] + b[i]);
}

// o = bf16(tgt + qpos), qpos = query_embed[:, 0:C] broadcast over batch
__global__ void k_addq_bf16(const float* __restrict__ tgt, const float* __restrict__ qe,
                            u16* __restrict__ o, long n)
{
    long i = blockIdx.x * (long)blockDim.x + threadIdx.x;
    if (i >= n) return;
    int c  = (int)(i % Cc);
    int nq = (int)((i / Cc) % NQn);
    o[i] = f2bfu(tgt[i] + qe[(long)nq * 2 * Cc + c]);
}

__global__ void k_poslev(const float* __restrict__ pf, const float* __restrict__ lev,
                         float* __restrict__ pos, long n)
{
    long i = blockIdx.x * (long)blockDim.x + threadIdx.x;
    if (i >= n) return;
    int c = (int)(i % Cc);
    int s = (int)((i / Cc) % STOT);
    int l = (s < 10000) ? 0 : (s < 12500) ? 1 : (s < 13125) ? 2 : 3;
    pos[i] = pf[i] + lev[l * Cc + c];
}

__global__ void k_inittgt(const float* __restrict__ qe, float* __restrict__ tgt, long n)
{
    long i = blockIdx.x * (long)blockDim.x + threadIdx.x;
    if (i >= n) return;
    int c  = (int)(i % Cc);
    int nq = (int)((i / Cc) % NQn);
    tgt[i] = qe[(long)nq * 2 * Cc + Cc + c];
}

__global__ void k_refp(const float* __restrict__ qe, const float* __restrict__ w,
                       const float* __restrict__ bb, float* __restrict__ refb)
{
    int i = blockIdx.x * blockDim.x + threadIdx.x;
    if (i >= NQn * 2) return;
    int nq = i >> 1, j = i & 1;
    float s = bb[j];
    for (int k = 0; k < Cc; ++k) s += qe[(long)nq * 2 * Cc + k] * w[j * Cc + k];
    float sig = 1.0f / (1.0f + __expf(-s));
    refb[(0 * NQn + nq) * 2 + j] = sig;
    refb[(1 * NQn + nq) * 2 + j] = sig;
}

// ---------------- softmax over 16 (L*P) per (token,head) ----------------
__global__ void k_softmax16(float* __restrict__ aw, long n)
{
    long i = blockIdx.x * (long)blockDim.x + threadIdx.x;
    if (i >= n) return;
    float* p = aw + i * 16;
    float mx = p[0];
    #pragma unroll
    for (int j = 1; j < 16; ++j) mx = fmaxf(mx, p[j]);
    float e[16], s = 0.0f;
    #pragma unroll
    for (int j = 0; j < 16; ++j) { e[j] = __expf(p[j] - mx); s += e[j]; }
    float inv = 1.0f / s;
    #pragma unroll
    for (int j = 0; j < 16; ++j) p[j] = e[j] * inv;
}

// ---------------- LayerNorm(out = LN(a + r) * g + be), C=256 ----------------
__global__ __launch_bounds__(256)
void k_ln(const float* __restrict__ a, const float* __restrict__ r,
          const float* __restrict__ g, const float* __restrict__ be,
          float* __restrict__ out)
{
    __shared__ float sh[256];
    int row = blockIdx.x, t = threadIdx.x;
    float v = a[(long)row * Cc + t] + r[(long)row * Cc + t];
    sh[t] = v; __syncthreads();
    for (int s = 128; s; s >>= 1) { if (t < s) sh[t] += sh[t + s]; __syncthreads(); }
    float mean = sh[0] * (1.0f / Cc); __syncthreads();
    float dv = v - mean;
    sh[t] = dv * dv; __syncthreads();
    for (int s = 128; s; s >>= 1) { if (t < s) sh[t] += sh[t + s]; __syncthreads(); }
    float var = sh[0] * (1.0f / Cc);
    out[(long)row * Cc + t] = dv * rsqrtf(var + 1e-5f) * g[t] + be[t];
}

// ---------------- MS deformable attention sampling ----------------
// grid (Lq, 2, B), block (32, 4): lane = channel d, threadIdx.y -> head sub-index.
__global__ void k_msda(const float* __restrict__ offs, const float* __restrict__ aw,
                       const float* __restrict__ val, const float* __restrict__ ref2,
                       float* __restrict__ out, int Lq)
{
    const int d = threadIdx.x;
    const int h = blockIdx.y * 4 + threadIdx.y;
    const int q = blockIdx.x;
    const int b = blockIdx.z;

    float rx, ry;
    if (ref2) {
        rx = ref2[((long)b * Lq + q) * 2 + 0];
        ry = ref2[((long)b * Lq + q) * 2 + 1];
    } else {
        int l0, idx;
        if      (q < 10000) { l0 = 0; idx = q; }
        else if (q < 12500) { l0 = 1; idx = q - 10000; }
        else if (q < 13125) { l0 = 2; idx = q - 12500; }
        else                { l0 = 3; idx = q - 13125; }
        int Wl = c_HW[l0];
        int y = idx / Wl, x = idx - y * Wl;
        rx = (x + 0.5f) / (float)Wl;
        ry = (y + 0.5f) / (float)Wl;
    }

    const long baseO = (((long)b * Lq + q) * HHn + h) * (LLn * PPn * 2);
    const long baseA = (((long)b * Lq + q) * HHn + h) * (LLn * PPn);
    float acc = 0.0f;

    for (int l = 0; l < LLn; ++l) {
        const int Wl = c_HW[l], st = c_ST[l];
        const float fW = (float)Wl;
        for (int p = 0; p < PPn; ++p) {
            float ox = offs[baseO + (l * PPn + p) * 2 + 0];
            float oy = offs[baseO + (l * PPn + p) * 2 + 1];
            float x = (rx + ox / fW) * fW - 0.5f;   // grid_sample, align_corners=False
            float y = (ry + oy / fW) * fW - 0.5f;
            float x0f = floorf(x), y0f = floorf(y);
            float wx = x - x0f, wy = y - y0f;
            int x0 = (int)x0f, y0 = (int)y0f;
            float a = aw[baseA + l * PPn + p];
            float s = 0.0f;
            #pragma unroll
            for (int cy = 0; cy < 2; ++cy)
                #pragma unroll
                for (int cx = 0; cx < 2; ++cx) {
                    int yi = y0 + cy, xi = x0 + cx;
                    float ok = (yi >= 0 && yi < Wl && xi >= 0 && xi < Wl) ? 1.0f : 0.0f;
                    int yc = min(max(yi, 0), Wl - 1);
                    int xc = min(max(xi, 0), Wl - 1);
                    long vidx = ((((long)b * STOT + st + yc * Wl + xc) * HHn + h) * DHn) + d;
                    float v = val[vidx] * ok;
                    float wgt = (cy ? wy : 1.0f - wy) * (cx ? wx : 1.0f - wx);
                    s += wgt * v;
                }
            acc += a * s;
        }
    }
    out[((long)b * Lq + q) * Cc + h * DHn + d] = acc;
}

// ---------------- fused decoder self-attention (NQ=300) ----------------
__global__ void k_selfattn(const float* __restrict__ qk, const float* __restrict__ vb,
                           float* __restrict__ out)
{
    __shared__ float sc[NQn];
    const int lane = threadIdx.x;
    const int q = blockIdx.x, h = blockIdx.y, b = blockIdx.z;
    const float scale = 0.1767766952966369f; // 1/sqrt(32)

    const float qd = qk[((long)b * NQn + q) * 512 + h * DHn + lane];
    for (int k = 0; k < NQn; ++k) {
        float v = qd * qk[((long)b * NQn + k) * 512 + 256 + h * DHn + lane];
        for (int o = 16; o; o >>= 1) v += __shfl_xor(v, o, 32);
        if (lane == 0) sc[k] = v * scale;
    }
    __syncthreads();

    float mx = -1e30f;
    for (int k = lane; k < NQn; k += 32) mx = fmaxf(mx, sc[k]);
    for (int o = 16; o; o >>= 1) mx = fmaxf(mx, __shfl_xor(mx, o, 32));
    float sm = 0.0f;
    for (int k = lane; k < NQn; k += 32) sm += __expf(sc[k] - mx);
    for (int o = 16; o; o >>= 1) sm += __shfl_xor(sm, o, 32);
    float inv = 1.0f / sm;

    float acc = 0.0f;
    for (int k = 0; k < NQn; ++k) {
        float p = __expf(sc[k] - mx) * inv;
        acc += p * vb[((long)b * NQn + k) * Cc + h * DHn + lane];
    }
    out[((long)b * NQn + q) * Cc + h * DHn + lane] = acc;
}

// ---------------- finalize: (tgt, init_ref, ref) concatenated ----------------
__global__ void k_final(const float* __restrict__ tgt, const float* __restrict__ refb,
                        float* __restrict__ out)
{
    int i = blockIdx.x * blockDim.x + threadIdx.x;
    const int ntgt = BB * NQn * Cc;
    if (i < ntgt) out[i] = tgt[i];
    if (i < BB * NQn * 2) {
        out[ntgt + i] = refb[i];
        out[ntgt + BB * NQn * 2 + i] = refb[i];
    }
}

// ---------------- host orchestration ----------------
extern "C" void kernel_launch(void* const* d_in, const int* in_sizes, int n_in,
                              void* d_out, int out_size, void* d_ws, size_t ws_size,
                              hipStream_t stream)
{
    const float* src   = (const float*)d_in[0];
    const float* posf  = (const float*)d_in[1];
    const float* qe    = (const float*)d_in[2];
    const float* lev   = (const float*)d_in[3];
    const float* refpw = (const float*)d_in[4];
    const float* refpb = (const float*)d_in[5];
    const float* eso_w = (const float*)d_in[6];  const float* eso_b = (const float*)d_in[7];
    const float* eaw_w = (const float*)d_in[8];  const float* eaw_b = (const float*)d_in[9];
    const float* evp_w = (const float*)d_in[10]; const float* evp_b = (const float*)d_in[11];
    const float* eop_w = (const float*)d_in[12]; const float* eop_b = (const float*)d_in[13];
    const float* en1_g = (const float*)d_in[14]; const float* en1_b = (const float*)d_in[15];
    const float* el1_w = (const float*)d_in[16]; const float* el1_b = (const float*)d_in[17];
    const float* el2_w = (const float*)d_in[18]; const float* el2_b = (const float*)d_in[19];
    const float* en2_g = (const float*)d_in[20]; const float* en2_b = (const float*)d_in[21];
    const float* dso_w = (const float*)d_in[22]; const float* dso_b = (const float*)d_in[23];
    const float* daw_w = (const float*)d_in[24]; const float* daw_b = (const float*)d_in[25];
    const float* dvp_w = (const float*)d_in[26]; const float* dvp_b = (const float*)d_in[27];
    const float* dop_w = (const float*)d_in[28]; const float* dop_b = (const float*)d_in[29];
    const float* dsain_w  = (const float*)d_in[30]; const float* dsain_b  = (const float*)d_in[31];
    const float* dsaout_w = (const float*)d_in[32]; const float* dsaout_b = (const float*)d_in[33];
    const float* dn1_g = (const float*)d_in[34]; const float* dn1_b = (const float*)d_in[35];
    const float* dn2_g = (const float*)d_in[36]; const float* dn2_b = (const float*)d_in[37];
    const float* dn3_g = (const float*)d_in[38]; const float* dn3_b = (const float*)d_in[39];
    const float* dl1_w = (const float*)d_in[40]; const float* dl1_b = (const float*)d_in[41];
    const float* dl2_w = (const float*)d_in[42]; const float* dl2_b = (const float*)d_in[43];

    const long SBC = (long)BB * STOT * Cc;   // 6,806,528
    const long NTC = (long)BB * NQn * Cc;    // 153,600

    char* cur = (char*)d_ws;
    auto fAl = [&](long n) { float* p = (float*)cur; cur += ((n * 4 + 63) & ~63L); return p; };
    auto hAl = [&](long n) { u16*   p = (u16*)cur;   cur += ((n * 2 + 63) & ~63L); return p; };

    float* pos  = fAl(SBC);
    float* mem  = fAl(SBC);
    float* qbuf = fAl(SBC);          // MSDA head-output
    float* val  = fAl(SBC);
    float* offs = fAl(SBC);          // also m2 / ffn-out
    float* awb  = fAl((long)BB * STOT * HHn * 16);
    float* qk   = fAl((long)BB * NQn * 512);
    float* vb   = fAl(NTC);
    float* sar  = fAl(NTC);
    float* dm2  = fAl(NTC);
    float* doff = fAl(NTC);
    float* dawb = fAl((long)BB * NQn * 128);
    float* dho  = fAl(NTC);
    float* refb = fAl((long)BB * NQn * 2);
    float* tgt  = fAl(NTC);

    u16* xb   = hAl(SBC);                       // bf16 GEMM input (K<=256 cases)
    u16* xmid = hAl((long)BB * STOT * FFn);     // bf16 FFN hidden
    u16* eso_bw = hAl((long)NLn * 256 * Cc);
    u16* eaw_bw = hAl((long)NLn * 128 * Cc);
    u16* evp_bw = hAl((long)NLn * Cc * Cc);
    u16* eop_bw = hAl((long)NLn * Cc * Cc);
    u16* el1_bw = hAl((long)NLn * FFn * Cc);
    u16* el2_bw = hAl((long)NLn * Cc * FFn);
    u16* dso_bw = hAl((long)NLn * 256 * Cc);
    u16* daw_bw = hAl((long)NLn * 128 * Cc);
    u16* dvp_bw = hAl((long)NLn * Cc * Cc);
    u16* dop_bw = hAl((long)NLn * Cc * Cc);
    u16* dsain_bw  = hAl((long)NLn * 768 * Cc);
    u16* dsaout_bw = hAl((long)NLn * Cc * Cc);
    u16* dl1_bw = hAl((long)NLn * FFn * Cc);
    u16* dl2_bw = hAl((long)NLn * Cc * FFn);

    auto cvt = [&](const float* x, u16* y, long n) {
        k_cvt<<<(int)((n / 4 + 255) / 256), 256, 0, stream>>>(x, y, n);
    };
    auto gemm = [&](const u16* X, const u16* Wm, const float* bias, void* Y,
                    int M, int N, int K, int relu, int obf) {
        dim3 g((N + 127) / 128, (M + 31) / 32);
        k_gemm_bf16<<<g, 128, 0, stream>>>(X, Wm, bias, Y, M, N, K, relu, obf);
    };

    const int M  = BB * STOT;   // 26588
    const int Md = BB * NQn;    // 600

    // ---- weights -> bf16 (once per launch) ----
    cvt(eso_w, eso_bw, (long)NLn * 256 * Cc);   cvt(eaw_w, eaw_bw, (long)NLn * 128 * Cc);
    cvt(evp_w, evp_bw, (long)NLn * Cc * Cc);    cvt(eop_w, eop_bw, (long)NLn * Cc * Cc);
    cvt(el1_w, el1_bw, (long)NLn * FFn * Cc);   cvt(el2_w, el2_bw, (long)NLn * Cc * FFn);
    cvt(dso_w, dso_bw, (long)NLn * 256 * Cc);   cvt(daw_w, daw_bw, (long)NLn * 128 * Cc);
    cvt(dvp_w, dvp_bw, (long)NLn * Cc * Cc);    cvt(dop_w, dop_bw, (long)NLn * Cc * Cc);
    cvt(dsain_w, dsain_bw, (long)NLn * 768 * Cc);
    cvt(dsaout_w, dsaout_bw, (long)NLn * Cc * Cc);
    cvt(dl1_w, dl1_bw, (long)NLn * FFn * Cc);   cvt(dl2_w, dl2_bw, (long)NLn * Cc * FFn);

    // ---- prep ----
    k_poslev<<<(int)((SBC + 255) / 256), 256, 0, stream>>>(posf, lev, pos, SBC);
    (void)hipMemcpyAsync(mem, src, SBC * sizeof(float), hipMemcpyDeviceToDevice, stream);

    // ---- encoder ----
    for (int i = 0; i < NLn; ++i) {
        cvt(mem, xb, SBC);
        gemm(xb, evp_bw + (long)i * Cc * Cc, evp_b + i * Cc, val, M, Cc, Cc, 0, 0);
        k_add_bf16<<<(int)((SBC + 255) / 256), 256, 0, stream>>>(mem, pos, xb, SBC);
        gemm(xb, eso_bw + (long)i * 256 * Cc, eso_b + i * 256, offs, M, 256, Cc, 0, 0);
        gemm(xb, eaw_bw + (long)i * 128 * Cc, eaw_b + i * 128, awb,  M, 128, Cc, 0, 0);
        long nsm = (long)M * HHn;
        k_softmax16<<<(int)((nsm + 255) / 256), 256, 0, stream>>>(awb, nsm);
        dim3 mg(STOT, 2, BB), mb(32, 4);
        k_msda<<<mg, mb, 0, stream>>>(offs, awb, val, nullptr, qbuf, STOT);
        cvt(qbuf, xb, SBC);
        gemm(xb, eop_bw + (long)i * Cc * Cc, eop_b + i * Cc, offs, M, Cc, Cc, 0, 0);
        k_ln<<<M, 256, 0, stream>>>(mem, offs, en1_g + i * Cc, en1_b + i * Cc, mem);
        cvt(mem, xb, SBC);
        gemm(xb,   el1_bw + (long)i * FFn * Cc, el1_b + i * FFn, xmid, M, FFn, Cc, 1, 1);
        gemm(xmid, el2_bw + (long)i * Cc * FFn, el2_b + i * Cc,  offs, M, Cc, FFn, 0, 0);
        k_ln<<<M, 256, 0, stream>>>(mem, offs, en2_g + i * Cc, en2_b + i * Cc, mem);
    }

    // ---- decoder prep ----
    k_inittgt<<<(int)((NTC + 255) / 256), 256, 0, stream>>>(qe, tgt, NTC);
    k_refp<<<(NQn * 2 + 63) / 64, 64, 0, stream>>>(qe, refpw, refpb, refb);

    // ---- decoder ----
    for (int i = 0; i < NLn; ++i) {
        k_addq_bf16<<<(int)((NTC + 255) / 256), 256, 0, stream>>>(tgt, qe, xb, NTC);
        gemm(xb, dsain_bw + (long)i * 768 * Cc, dsain_b + i * 768, qk, Md, 512, Cc, 0, 0);
        cvt(tgt, xb, NTC);
        gemm(xb, dsain_bw + (long)i * 768 * Cc + 512 * Cc, dsain_b + i * 768 + 512, vb, Md, Cc, Cc, 0, 0);
        dim3 ag(NQn, HHn, BB);
        k_selfattn<<<ag, 32, 0, stream>>>(qk, vb, sar);
        cvt(sar, xb, NTC);
        gemm(xb, dsaout_bw + (long)i * Cc * Cc, dsaout_b + i * Cc, dm2, Md, Cc, Cc, 0, 0);
        k_ln<<<Md, 256, 0, stream>>>(tgt, dm2, dn2_g + i * Cc, dn2_b + i * Cc, tgt);

        k_addq_bf16<<<(int)((NTC + 255) / 256), 256, 0, stream>>>(tgt, qe, xb, NTC);
        gemm(xb, dso_bw + (long)i * 256 * Cc, dso_b + i * 256, doff, Md, 256, Cc, 0, 0);
        gemm(xb, daw_bw + (long)i * 128 * Cc, daw_b + i * 128, dawb, Md, 128, Cc, 0, 0);
        long nsm = (long)Md * HHn;
        k_softmax16<<<(int)((nsm + 255) / 256), 256, 0, stream>>>(dawb, nsm);
        cvt(mem, xb, SBC);
        gemm(xb, dvp_bw + (long)i * Cc * Cc, dvp_b + i * Cc, val, M, Cc, Cc, 0, 0);
        dim3 mg(NQn, 2, BB), mb(32, 4);
        k_msda<<<mg, mb, 0, stream>>>(doff, dawb, val, refb, dho, NQn);
        cvt(dho, xb, NTC);
        gemm(xb, dop_bw + (long)i * Cc * Cc, dop_b + i * Cc, dm2, Md, Cc, Cc, 0, 0);
        k_ln<<<Md, 256, 0, stream>>>(tgt, dm2, dn1_g + i * Cc, dn1_b + i * Cc, tgt);

        cvt(tgt, xb, NTC);
        gemm(xb,   dl1_bw + (long)i * FFn * Cc, dl1_b + i * FFn, xmid, Md, FFn, Cc, 1, 1);
        gemm(xmid, dl2_bw + (long)i * Cc * FFn, dl2_b + i * Cc,  dm2,  Md, Cc, FFn, 0, 0);
        k_ln<<<Md, 256, 0, stream>>>(tgt, dm2, dn3_g + i * Cc, dn3_b + i * Cc, tgt);
    }

    k_final<<<(BB * NQn * Cc + 255) / 256, 256, 0, stream>>>(tgt, refb, (float*)d_out);
    (void)in_sizes; (void)n_in; (void)out_size; (void)ws_size;
}
